// get_point_correspondences_28613072126431
// MI455X (gfx1250) — matrix operand order repaired and checked by reading, not compiled, not run on hardware
//
#include <hip/hip_runtime.h>
#include <math.h>
#include <stdint.h>

typedef __attribute__((ext_vector_type(16))) _Float16 v16h;
typedef __attribute__((ext_vector_type(8)))  float    v8f;

#define B_  4
#define D_  256
#define N_  4096
#define M_  4096
#define TM  16            // m-tile width
#define NT  (M_/TM)       // 256 tiles
#define LSTRIDE 272       // halves per m-row in LDS (keeps 32B align, near conflict-free)
#define LOG2E 1.44269504088896340736f

__device__ __forceinline__ float ex2(float x) { return __builtin_amdgcn_exp2f(x); }

// ---------------------------------------------------------------------------
// yy[b][m] = sum_d (f16(tgt_emb[b][d][m]))^2, pre-scaled by log2(e) so the
// main kernel works directly in base-2 exponent space.
// ---------------------------------------------------------------------------
__global__ __launch_bounds__(256) void yy_kernel(const float* __restrict__ tgt_emb,
                                                 float* __restrict__ yy)
{
  const int idx = blockIdx.x * 256 + threadIdx.x;     // 0 .. B*M-1
  const int b = idx / M_, m = idx % M_;
  const float* p = tgt_emb + (size_t)b * D_ * M_ + m;
  float s = 0.f;
#pragma unroll 8
  for (int d = 0; d < D_; ++d) {
    _Float16 h = (_Float16)p[(size_t)d * M_];
    float f = (float)h;
    s += f * f;
  }
  yy[idx] = s * LOG2E;
}

// ---------------------------------------------------------------------------
// One-time transpose + f32->f16 convert: tgt_emb [B][D][M] -> k16 [B][M][D].
// Coalesced reads over m, LDS 16x256 tile transpose, 32B coalesced writes.
// ---------------------------------------------------------------------------
__global__ __launch_bounds__(256) void cvt_kernel(const float* __restrict__ tgt_emb,
                                                  _Float16* __restrict__ k16)
{
  __shared__ __align__(32) _Float16 lds[TM * LSTRIDE];
  const int t  = threadIdx.x;                   // = d row
  const int b  = blockIdx.x / (M_ / TM);
  const int m0 = (blockIdx.x % (M_ / TM)) * TM;

  const float4* p4 = (const float4*)(tgt_emb + ((size_t)b * D_ + t) * M_ + m0);
  float4 f0 = p4[0], f1 = p4[1], f2 = p4[2], f3 = p4[3];
  float vals[16] = { f0.x, f0.y, f0.z, f0.w, f1.x, f1.y, f1.z, f1.w,
                     f2.x, f2.y, f2.z, f2.w, f3.x, f3.y, f3.z, f3.w };
#pragma unroll
  for (int mm = 0; mm < 16; ++mm) lds[mm * LSTRIDE + t] = (_Float16)vals[mm];
  __syncthreads();

  const int ml = t >> 4, dblk = (t & 15) * 16;
  v16h v = *(const v16h*)&lds[ml * LSTRIDE + dblk];
  *(v16h*)&k16[((size_t)b * M_ + m0 + ml) * D_ + dblk] = v;
}

// ---------------------------------------------------------------------------
// Fused distance -> per-lane online softmax -> 3-channel V accumulation.
// Block = 8 waves; each wave owns 16 src rows (A fragments resident in VGPRs).
// USE16=1: K-tiles staged via global_load_async_to_lds_b128 (ASYNCcnt) from
// the pre-converted f16 K; USE16=0: register staging fallback.
// ---------------------------------------------------------------------------
template<int USE16>
__global__ __launch_bounds__(256) void corr_kernel(const float* __restrict__ tgt,
                                                   const float* __restrict__ src_emb,
                                                   const float* __restrict__ tgt_emb,
                                                   const _Float16* __restrict__ k16,
                                                   const float* __restrict__ yyg,
                                                   float* __restrict__ out)
{
  __shared__ __align__(32) _Float16 lds[2][TM * LSTRIDE];

  const int tid  = threadIdx.x;
  const int wave = tid >> 5;
  const int lane = tid & 31;
  const int hs   = lane >> 4;
  const int r    = lane & 15;

  const int b    = blockIdx.x >> 5;          // 32 blocks per batch
  const int nblk = blockIdx.x & 31;
  const int n0   = nblk * 128 + wave * 16;

  // ---- A fragments: 16 rows x 256 d of src_emb as f16, kept resident ------
  const float* se = src_emb + (size_t)b * D_ * N_;
  v16h afrag[8];
  float xxp = 0.f;
#pragma unroll
  for (int kc = 0; kc < 8; ++kc) {
    const int kb = kc * 32;
#pragma unroll
    for (int j = 0; j < 8; ++j) {
      float x0 = se[(size_t)(kb + hs * 16 + j) * N_ + n0 + r];
      float x1 = se[(size_t)(kb + hs * 16 + 8 + j) * N_ + n0 + r];
      _Float16 h0 = (_Float16)x0, h1 = (_Float16)x1;
      afrag[kc][j]     = h0;
      afrag[kc][8 + j] = h1;
      float f0 = (float)h0, f1 = (float)h1;
      xxp += f0 * f0 + f1 * f1;
    }
  }
  float xxr = xxp + __shfl_xor(xxp, 16, 32);          // lane holds ||q_r||^2
  float xs[8];                                        // per-C-row xx * log2e
#pragma unroll
  for (int v = 0; v < 8; ++v) xs[v] = __shfl(xxr, v + 8 * hs, 32) * LOG2E;

  // ---- per-lane streaming softmax state (columns m ≡ r mod 16) ------------
  float mrun[8], rsum[8], a0[8], a1[8], a2[8];
#pragma unroll
  for (int v = 0; v < 8; ++v) {
    mrun[v] = -INFINITY; rsum[v] = 0.f; a0[v] = 0.f; a1[v] = 0.f; a2[v] = 0.f;
  }

  const float* tg0  = tgt + (size_t)b * 3 * M_;
  const float* tg1  = tg0 + M_;
  const float* tg2  = tg1 + M_;
  const float* yrow = yyg + (size_t)b * M_;
  const float* teb  = tgt_emb + ((size_t)b * D_ + tid) * M_;   // f32 fallback row

  const int ml = tid >> 4, dblk = (tid & 15) * 16;             // async staging coords

  // per-thread staging: 32 bytes of the f16 K-tile, async copied into LDS
  auto stage_async = [&](int mtile, int nb) {
    const _Float16* g = k16 + ((size_t)b * M_ + (size_t)mtile * TM + ml) * D_ + dblk;
    uint64_t ga = (uint64_t)(uintptr_t)g;
    uint32_t lo = (uint32_t)(uintptr_t)&lds[nb][ml * LSTRIDE + dblk];
    asm volatile("global_load_async_to_lds_b128 %0, %1, off\n\t"
                 "global_load_async_to_lds_b128 %0, %1, off offset:16"
                 :: "v"(lo), "v"(ga) : "memory");
  };
  auto stage_f32 = [&](int mtile, int nb) {                    // fallback path
    const float4* p4 = (const float4*)(teb + (size_t)mtile * TM);
    float4 f0 = p4[0], f1 = p4[1], f2 = p4[2], f3 = p4[3];
    float vals[16] = { f0.x, f0.y, f0.z, f0.w, f1.x, f1.y, f1.z, f1.w,
                       f2.x, f2.y, f2.z, f2.w, f3.x, f3.y, f3.z, f3.w };
    _Float16* dst = &lds[nb][0];
#pragma unroll
    for (int mm = 0; mm < 16; ++mm) dst[mm * LSTRIDE + tid] = (_Float16)vals[mm];
  };

  if (USE16) { stage_async(0, 0); asm volatile("s_wait_asynccnt 0x0" ::: "memory"); }
  else       { stage_f32(0, 0); }
  __syncthreads();

  const float C2 = 2.0f * LOG2E;
  int buf = 0;
  for (int mt = 0; mt < NT; ++mt) {
    const int m0  = mt * TM;
    const int mtn = (mt + 1 < NT) ? (mt + 1) : mt;

    if (USE16) stage_async(mtn, buf ^ 1);    // fire-and-forget into other buffer

    // -- preload all 8 B fragments, then chain 8 WMMAs ----------------------
    const _Float16* bb = &lds[buf][r * LSTRIDE + hs * 16];
    v16h bfrag[8];
#pragma unroll
    for (int kc = 0; kc < 8; ++kc) bfrag[kc] = *(const v16h*)(bb + kc * 32);

    v8f acc = {};
#pragma unroll
    for (int kc = 0; kc < 8; ++kc)
      acc = __builtin_amdgcn_wmma_f32_16x16x32_f16(false, afrag[kc], false, bfrag[kc],
                                                   (short)0, acc, false, false);

    if (!USE16) stage_f32(mtn, buf ^ 1);

    // -- per-lane online softmax in base-2 (no cross-lane ops in hot loop) --
    const float nyy = -yrow[m0 + r];         // already * log2e
    const float t0  = tg0[m0 + r];
    const float t1  = tg1[m0 + r];
    const float t2  = tg2[m0 + r];
    float cterm[8];
#pragma unroll
    for (int v = 0; v < 8; ++v) cterm[v] = nyy - xs[v];
#pragma unroll
    for (int v = 0; v < 8; ++v) {
      float s2 = fmaf(acc[v], C2, cterm[v]);      // log2e * (2*dot - xx - yy)
      float nm = fmaxf(mrun[v], s2);
      float sc = ex2(mrun[v] - nm);               // -inf first time -> 0
      float p  = ex2(s2 - nm);
      rsum[v] = fmaf(rsum[v], sc, p);
      a0[v]   = fmaf(a0[v],   sc, p * t0);
      a1[v]   = fmaf(a1[v],   sc, p * t1);
      a2[v]   = fmaf(a2[v],   sc, p * t2);
      mrun[v] = nm;
    }

    if (USE16) asm volatile("s_wait_asynccnt 0x0" ::: "memory");
    __syncthreads();
    buf ^= 1;
  }

  // ---- merge 16 per-lane partials per row (different maxes), then store ---
#pragma unroll
  for (int v = 0; v < 8; ++v) {
    float m = mrun[v], s = rsum[v], b0 = a0[v], b1 = a1[v], b2 = a2[v];
#pragma unroll
    for (int mask = 1; mask <= 8; mask <<= 1) {
      float mo = __shfl_xor(m,  mask, 32);
      float so = __shfl_xor(s,  mask, 32);
      float o0 = __shfl_xor(b0, mask, 32);
      float o1 = __shfl_xor(b1, mask, 32);
      float o2 = __shfl_xor(b2, mask, 32);
      float nm = fmaxf(m, mo);
      float es = ex2(m - nm), eo = ex2(mo - nm);
      s  = fmaf(s,  es, so * eo);
      b0 = fmaf(b0, es, o0 * eo);
      b1 = fmaf(b1, es, o1 * eo);
      b2 = fmaf(b2, es, o2 * eo);
      m  = nm;
    }
    if (r == v) {
      const int n = n0 + 8 * hs + v;
      const float inv = 1.f / s;                  // >= 1, never 0
      out[((size_t)b * 3 + 0) * N_ + n] = b0 * inv;
      out[((size_t)b * 3 + 1) * N_ + n] = b1 * inv;
      out[((size_t)b * 3 + 2) * N_ + n] = b2 * inv;
    }
  }
}

extern "C" void kernel_launch(void* const* d_in, const int* in_sizes, int n_in,
                              void* d_out, int out_size, void* d_ws, size_t ws_size,
                              hipStream_t stream) {
  (void)in_sizes; (void)n_in; (void)out_size;
  const float* tgt     = (const float*)d_in[1];   // d_in[0] (src) unused by reference
  const float* src_emb = (const float*)d_in[2];
  const float* tgt_emb = (const float*)d_in[3];
  float* out = (float*)d_out;

  float*    yy  = (float*)d_ws;                                   // 64 KB
  _Float16* k16 = (_Float16*)((char*)d_ws + (size_t)B_ * M_ * 4); // 8 MB, 32B-aligned
  const size_t need = (size_t)B_ * M_ * 4 + (size_t)B_ * M_ * D_ * 2;
  const bool use16 = (ws_size >= need);

  yy_kernel<<<(B_ * M_) / 256, 256, 0, stream>>>(tgt_emb, yy);
  if (use16) {
    cvt_kernel<<<B_ * (M_ / TM), 256, 0, stream>>>(tgt_emb, k16);
    corr_kernel<1><<<B_ * (N_ / 128), 256, 0, stream>>>(tgt, src_emb, tgt_emb, k16, yy, out);
  } else {
    corr_kernel<0><<<B_ * (N_ / 128), 256, 0, stream>>>(tgt, src_emb, tgt_emb, k16, yy, out);
  }
}
